// HMamba_37383395344516
// MI455X (gfx1250) — compile-verified, matrix operand order checked by reading, not run on hardware
//
#include <hip/hip_runtime.h>
#include <hip/hip_bf16.h>
#include <hip/hip_fp16.h>

typedef __attribute__((ext_vector_type(16))) _Float16 v16h;
typedef __attribute__((ext_vector_type(8)))  _Float16 v8h;
typedef __attribute__((ext_vector_type(8)))  float    v8f;

#define NTOK  32768
#define TBLK  128      // attention block size
#define NH    8
#define HDIM  64
#define NLAY  2
#define DIN   128      // mamba inner dim
#define DST   16       // mamba state dim
#define CDIM  6
#define CHUNK 128
#define NCHUNK (NTOK / CHUNK)

// ---------------------------------------------------------------------------
// WMMA fragment helpers (layouts per CDNA5 ISA 7.12.2, wave32).
// A-fragment f16 16x32: lane m=lane&15, kh=lane>>4; VGPR v<4 holds
// k = 8*kh + 2v,2v+1 (contiguous 8 f16), VGPR v>=4 holds k = 16+8*kh+...
// => fragment == two 16-byte contiguous loads per lane.
// ---------------------------------------------------------------------------
__device__ __forceinline__ v8f wmma_f16(v16h a, v16h b, v8f c) {
    return __builtin_amdgcn_wmma_f32_16x16x32_f16(false, a, false, b, (short)0, c, false, false);
}

// A fragment: element (m,k) = A[(m0+m)*lda + k0+k], row-major A. 16B-aligned rows.
__device__ __forceinline__ v16h load_a_frag(const _Float16* A, int lda, int m0, int k0, int lane) {
    int m  = m0 + (lane & 15);
    int kh = lane >> 4;
    const _Float16* p = A + (long)m * lda + k0 + 8 * kh;
    v8h lo = *(const v8h*)p;          // VGPR 0..3
    v8h hi = *(const v8h*)(p + 16);   // VGPR 4..7
    return __builtin_shufflevector(lo, hi, 0,1,2,3,4,5,6,7,8,9,10,11,12,13,14,15);
}

// B fragment from transposed storage: element (k,n) = T[(n0+n)*ldt + k0+k].
// Same lane striping as A but indexed by n; also two 16-byte loads.
__device__ __forceinline__ v16h load_bT_frag(const _Float16* T, int ldt, int k0, int n0, int lane) {
    int n  = n0 + (lane & 15);
    int kh = lane >> 4;
    const _Float16* p = T + (long)n * ldt + k0 + 8 * kh;
    v8h lo = *(const v8h*)p;
    v8h hi = *(const v8h*)(p + 16);
    return __builtin_shufflevector(lo, hi, 0,1,2,3,4,5,6,7,8,9,10,11,12,13,14,15);
}

// ---------------------------------------------------------------------------
// Sorting (bitonic on u64 keys: [coord bits | index])
// ---------------------------------------------------------------------------
__global__ void k_sort_init(const float* __restrict__ coords, unsigned long long* __restrict__ keys) {
    int i = blockIdx.x * 256 + threadIdx.x;
    if (i < NTOK) {
        unsigned int kb = __float_as_uint(coords[(long)i * CDIM]);  // uniform [0,1): bits sortable
        keys[i] = ((unsigned long long)kb << 32) | (unsigned int)i;
    }
}

__global__ void k_sort_stage(unsigned long long* __restrict__ keys, int j, int k) {
    int i   = blockIdx.x * 256 + threadIdx.x;
    int ixj = i ^ j;
    if (ixj > i && ixj < NTOK && i < NTOK) {
        unsigned long long a = keys[i], b = keys[ixj];
        bool asc = ((i & k) == 0);
        if (asc ? (a > b) : (a < b)) { keys[i] = b; keys[ixj] = a; }
    }
}

__global__ void k_sort_finish(const unsigned long long* __restrict__ keys, int* __restrict__ order) {
    int i = blockIdx.x * 256 + threadIdx.x;
    if (i < NTOK) order[i] = (int)(keys[i] & 0xffffffffu);
}

// ---------------------------------------------------------------------------
// Gather + LayerNorm (ln1) -> f16, plus gathered coords[1:6]
// ---------------------------------------------------------------------------
__global__ void k_gather_ln(const float* __restrict__ x, const float* __restrict__ coords,
                            const int* __restrict__ order,
                            const float* __restrict__ g, const float* __restrict__ b,
                            _Float16* __restrict__ xn16, float* __restrict__ cs5) {
    int row = blockIdx.x, t = threadIdx.x;   // blockDim = 64
    __shared__ float red[64];
    __shared__ int gsh;
    if (t == 0) gsh = order[row];
    __syncthreads();
    int gidx = gsh;
    float v = x[(long)gidx * HDIM + t];
    red[t] = v; __syncthreads();
    for (int s = 32; s > 0; s >>= 1) { if (t < s) red[t] += red[t + s]; __syncthreads(); }
    float mu = red[0] * (1.0f / 64.0f);
    __syncthreads();
    float d = v - mu;
    red[t] = d * d; __syncthreads();
    for (int s = 32; s > 0; s >>= 1) { if (t < s) red[t] += red[t + s]; __syncthreads(); }
    float var = red[0] * (1.0f / 64.0f);
    float r = rsqrtf(var + 1e-5f);
    xn16[(long)row * HDIM + t] = (_Float16)(d * r * g[t] + b[t]);
    if (t < 5) cs5[(long)row * 5 + t] = coords[(long)gidx * CDIM + 1 + t];
}

// Generic LayerNorm with optional residual accumulate, f16 and/or f32 output.
__global__ void k_ln(const float* __restrict__ src, float* __restrict__ resid,
                     const float* __restrict__ g, const float* __restrict__ b,
                     _Float16* __restrict__ out16, float* __restrict__ out32) {
    int row = blockIdx.x, t = threadIdx.x;   // blockDim = 64
    __shared__ float red[64];
    float v = src[(long)row * HDIM + t];
    if (resid) { v += resid[(long)row * HDIM + t]; resid[(long)row * HDIM + t] = v; }
    red[t] = v; __syncthreads();
    for (int s = 32; s > 0; s >>= 1) { if (t < s) red[t] += red[t + s]; __syncthreads(); }
    float mu = red[0] * (1.0f / 64.0f);
    __syncthreads();
    float d = v - mu;
    red[t] = d * d; __syncthreads();
    for (int s = 32; s > 0; s >>= 1) { if (t < s) red[t] += red[t + s]; __syncthreads(); }
    float var = red[0] * (1.0f / 64.0f);
    float r = rsqrtf(var + 1e-5f);
    float o = d * r * g[t] + b[t];
    if (out16) out16[(long)row * HDIM + t] = (_Float16)o;
    if (out32) out32[(long)row * HDIM + t] = o;
}

// ---------------------------------------------------------------------------
// Weight prep: f32 -> f16 with transpose (B matrices stored [Nc][K])
// ---------------------------------------------------------------------------
__global__ void k_f16T(const float* __restrict__ s, _Float16* __restrict__ d, int K, int Nc) {
    int i = blockIdx.x * 256 + threadIdx.x;   // over K*Nc
    if (i < K * Nc) {
        int k = i / Nc, n = i % Nc;
        d[(long)n * K + k] = (_Float16)s[i];
    }
}

// x_proj: s 128x36 -> d (transposed, padded) 64x128; d[n][r] = n<36 ? s[r][n] : 0
__global__ void k_pad_xprojT(const float* __restrict__ s, _Float16* __restrict__ d) {
    int i = blockIdx.x * 256 + threadIdx.x;
    if (i < 64 * 128) {
        int n = i >> 7, r = i & 127;
        d[i] = (n < 36) ? (_Float16)s[r * 36 + n] : (_Float16)0.0f;
    }
}

__global__ void k_rpe_prep(const float* __restrict__ rpe_w, const float* __restrict__ rpe_b,
                           float* __restrict__ effw, float* __restrict__ biasb) {
    int t = threadIdx.x;  // single block of 64
    if (t < 40) {
        int f = t / 8, h = t % 8;
        float acc = 0.f;
        for (int w = 0; w < 8; ++w)
            for (int d = 0; d < 64; ++d)
                acc += rpe_w[(long)(w * 5 + f) * 512 + h * 64 + d];
        effw[f * 8 + h] = acc * (1.0f / 64.0f);
    }
    if (t < 8) {
        float a = 0.f;
        for (int d = 0; d < 64; ++d) a += rpe_b[t * 64 + d];
        biasb[t] = a * (1.0f / 64.0f);
    }
}

// ---------------------------------------------------------------------------
// WMMA GEMM: C[MxNc] = A[MxK] * Bt^T  (Bt stored [Nc][K] f16).
// Workgroup = 256 thr = 8 waves -> 64x32 tile; wave = 16x16 tile.
// Software-pipelined: fragments for tile k+1 are loaded before the WMMA of
// tile k, so vmem stays in flight across the matrix op. Prefetch is
// speculative (ISA: invalid prefetch addresses silently dropped) -> no guard.
// ---------------------------------------------------------------------------
__device__ __forceinline__ v8f gemm_core(const _Float16* __restrict__ A,
                                         const _Float16* __restrict__ Bt,
                                         int K, int m0, int n0, int lane) {
    v8f acc = {};
    v16h a = load_a_frag(A, K, m0, 0, lane);
    v16h b = load_bT_frag(Bt, K, 0, n0, lane);
    for (int k0 = 32; k0 < K; k0 += 32) {
        __builtin_prefetch(A + (long)(m0 + (lane & 15)) * K + k0 + 32, 0, 3);
        v16h an = load_a_frag(A, K, m0, k0, lane);
        v16h bn = load_bT_frag(Bt, K, k0, n0, lane);
        acc = wmma_f16(a, b, acc);
        a = an; b = bn;
    }
    acc = wmma_f16(a, b, acc);
    return acc;
}

__global__ void __launch_bounds__(256)
k_gemm16(const _Float16* __restrict__ A, const _Float16* __restrict__ Bt,
         _Float16* __restrict__ C16, int K, int Nc, int nbn) {
    int bid = blockIdx.x;
    int bm = bid / nbn, bn = bid % nbn;
    int tid = threadIdx.x, lane = tid & 31, w = tid >> 5;
    int m0 = bm * 64 + (w >> 1) * 16;
    int n0 = bn * 32 + (w & 1) * 16;
    v8f acc = gemm_core(A, Bt, K, m0, n0, lane);
    int n = n0 + (lane & 15), mh = lane >> 4;
#pragma unroll
    for (int v = 0; v < 8; ++v)
        C16[(long)(m0 + 8 * mh + v) * Nc + n] = (_Float16)acc[v];
}

__global__ void __launch_bounds__(256)
k_gemm32(const _Float16* __restrict__ A, const _Float16* __restrict__ Bt,
         float* __restrict__ C32, int K, int Nc, int nbn) {
    int bid = blockIdx.x;
    int bm = bid / nbn, bn = bid % nbn;
    int tid = threadIdx.x, lane = tid & 31, w = tid >> 5;
    int m0 = bm * 64 + (w >> 1) * 16;
    int n0 = bn * 32 + (w & 1) * 16;
    v8f acc = gemm_core(A, Bt, K, m0, n0, lane);
    int n = n0 + (lane & 15), mh = lane >> 4;
#pragma unroll
    for (int v = 0; v < 8; ++v)
        C32[(long)(m0 + 8 * mh + v) * Nc + n] = acc[v];
}

// ---------------------------------------------------------------------------
// Attention: one workgroup per (block, head, row-quarter of 32 rows)
// ---------------------------------------------------------------------------
__global__ void __launch_bounds__(256)
k_attn(const _Float16* __restrict__ q16, const _Float16* __restrict__ k16,
       const _Float16* __restrict__ v16, const float* __restrict__ cs5,
       const float* __restrict__ effw, const float* __restrict__ biasb,
       _Float16* __restrict__ o16) {
    int bid = blockIdx.x;
    int qr  = bid & 3;
    int h   = (bid >> 2) & 7;
    int blk = bid >> 5;
    int tid = threadIdx.x, lane = tid & 31, w = tid >> 5;

    __shared__ __attribute__((aligned(16))) _Float16 kt [TBLK * HDIM];  // [tok][d], 16 KB
    __shared__ __attribute__((aligned(16))) _Float16 vtT[HDIM * TBLK];  // [d][tok], 16 KB
    __shared__ __attribute__((aligned(16))) float    Sb [32 * TBLK];    // 16 KB
    __shared__ __attribute__((aligned(16))) _Float16 Pb [32 * TBLK];    // 8 KB
    __shared__ float c5[TBLK * 5];                                      // 2.5 KB

    int base = blk * TBLK;
    // stage K (row-major) and V (transposed) with 16-byte global loads
    for (int i = tid; i < (TBLK * HDIM) / 8; i += 256) {
        int r = i >> 3, c8 = (i & 7) * 8;
        v8h kv = *(const v8h*)(k16 + (long)(base + r) * 512 + h * 64 + c8);
        *(v8h*)(kt + r * HDIM + c8) = kv;
        v8h vv = *(const v8h*)(v16 + (long)(base + r) * 512 + h * 64 + c8);
#pragma unroll
        for (int c = 0; c < 8; ++c) vtT[(c8 + c) * TBLK + r] = vv[c];
    }
    for (int i = tid; i < TBLK * 5; i += 256) c5[i] = cs5[(long)base * 5 + i];
    __syncthreads();

    int r0 = qr * 32;
    // ---- S = Q K^T * 1/sqrt(64) ----
    {
        int mt  = w >> 2;                 // 0..1
        int m0g = base + r0 + mt * 16;    // global q row
        v16h a0 = load_a_frag(q16, 512, m0g, h * 64 + 0, lane);
        v16h a1 = load_a_frag(q16, 512, m0g, h * 64 + 32, lane);
#pragma unroll
        for (int nn = 0; nn < 2; ++nn) {
            int nt = (w & 3) + nn * 4;    // 0..7
            int n0 = nt * 16;
            v16h b0 = load_bT_frag(kt, HDIM, 0,  n0, lane);
            v16h b1 = load_bT_frag(kt, HDIM, 32, n0, lane);
            v8f acc = {};
            acc = wmma_f16(a0, b0, acc);
            acc = wmma_f16(a1, b1, acc);
            int n = n0 + (lane & 15), mh = lane >> 4;
#pragma unroll
            for (int v = 0; v < 8; ++v)
                Sb[(mt * 16 + 8 * mh + v) * TBLK + n] = acc[v] * 0.125f;
        }
    }
    __syncthreads();
    // ---- RPE bias ----
    float ew[5];
#pragma unroll
    for (int f = 0; f < 5; ++f) ew[f] = effw[f * 8 + h];
    float bb = biasb[h];
    for (int i = tid; i < 32 * TBLK; i += 256) {
        int il = i >> 7, j = i & 127;
        int ig = r0 + il;
        float bias = bb;
#pragma unroll
        for (int f = 0; f < 5; ++f) bias += fabsf(c5[ig * 5 + f] - c5[j * 5 + f]) * ew[f];
        Sb[i] += bias;
    }
    __syncthreads();
    // ---- softmax over 128 cols, rows handled by threads 0..31 ----
    if (tid < 32) {
        float mx = -1e30f;
        for (int j = 0; j < TBLK; ++j) mx = fmaxf(mx, Sb[tid * TBLK + j]);
        float sum = 0.f;
        for (int j = 0; j < TBLK; ++j) { float e = __expf(Sb[tid * TBLK + j] - mx); Sb[tid * TBLK + j] = e; sum += e; }
        float inv = 1.0f / sum;
        for (int j = 0; j < TBLK; ++j) Pb[tid * TBLK + j] = (_Float16)(Sb[tid * TBLK + j] * inv);
    }
    __syncthreads();
    // ---- O = P V : 2 mtiles x 4 d-tiles, software-pipelined over K=128 ----
    {
        int mt = w >> 2, nt = w & 3;
        int m0l = mt * 16, n0 = nt * 16;
        v8f acc = {};
        v16h a = load_a_frag(Pb, TBLK, m0l, 0, lane);
        v16h b = load_bT_frag(vtT, TBLK, 0, n0, lane);
#pragma unroll
        for (int k0 = 32; k0 < TBLK; k0 += 32) {
            v16h an = load_a_frag(Pb, TBLK, m0l, k0, lane);
            v16h bn = load_bT_frag(vtT, TBLK, k0, n0, lane);
            acc = wmma_f16(a, b, acc);
            a = an; b = bn;
        }
        acc = wmma_f16(a, b, acc);
        int n = n0 + (lane & 15), mh = lane >> 4;
#pragma unroll
        for (int v = 0; v < 8; ++v) {
            int m = r0 + m0l + 8 * mh + v;
            o16[(long)(base + m) * 512 + h * 64 + n] = (_Float16)acc[v];
        }
    }
}

// scatter: hidden[order[j]] = tmp[j]    (implements hidden = (o@W)[inv])
__global__ void k_scatter(const float* __restrict__ tmp, const int* __restrict__ order,
                          float* __restrict__ hidden) {
    int i = blockIdx.x * 256 + threadIdx.x;
    if (i < NTOK * HDIM) {
        int row = i >> 6, c = i & 63;
        hidden[(long)order[row] * HDIM + c] = tmp[i];
    }
}

// ---------------------------------------------------------------------------
// Mamba pieces
// ---------------------------------------------------------------------------
// causal depthwise conv4 + SiLU: reads xi = xz[:, 0:128]
__global__ void k_conv(const float* __restrict__ xz, const float* __restrict__ cw,
                       const float* __restrict__ cb, float* __restrict__ xc,
                       _Float16* __restrict__ xc16) {
    int i = blockIdx.x * 256 + threadIdx.x;
    if (i >= NTOK * DIN) return;
    int n = i >> 7, d = i & 127;
    float acc = cb[d];
#pragma unroll
    for (int kk = 0; kk < 4; ++kk) {
        int nn = n + kk - 3;
        float xv = (nn >= 0) ? xz[(long)nn * 256 + d] : 0.f;
        acc += xv * cw[d * 4 + kk];
    }
    float s = acc / (1.0f + __expf(-acc));
    xc[i] = s;
    xc16[i] = (_Float16)s;
}

// dt = softplus(dtr @ dtw + dtb); dtr = dbl[:, 0:4] (dbl stride 64)
__global__ void k_dt(const float* __restrict__ dbl, const float* __restrict__ dtw,
                     const float* __restrict__ dtb, float* __restrict__ dt) {
    int i = blockIdx.x * 256 + threadIdx.x;
    if (i >= NTOK * DIN) return;
    int n = i >> 7, d = i & 127;
    float a = dtb[d];
#pragma unroll
    for (int r = 0; r < 4; ++r) a += dbl[(long)n * 64 + r] * dtw[r * DIN + d];
    dt[i] = (a > 20.f) ? a : log1pf(__expf(a));
}

// phase 1: per-chunk local scan -> carries (aprod, hend)
__global__ void __launch_bounds__(256)
k_scan1(const float* __restrict__ dt, const float* __restrict__ dbl,
        const float* __restrict__ xc, const float* __restrict__ a_log,
        float* __restrict__ aprod_o, float* __restrict__ hend_o) {
    int bid = blockIdx.x;            // NCHUNK * 8
    int chunk = bid >> 3, dg = bid & 7;
    int tid = threadIdx.x;
    int s = tid & 15, dl = tid >> 4;
    int d = dg * 16 + dl;
    __shared__ float dtl[CHUNK * 16], xcl[CHUNK * 16], Bl[CHUNK * 16];
    int n0 = chunk * CHUNK;
    for (int i = tid; i < CHUNK * 16; i += 256) {
        int n = i >> 4, c = i & 15;
        dtl[i] = dt[(long)(n0 + n) * DIN + dg * 16 + c];
        xcl[i] = xc[(long)(n0 + n) * DIN + dg * 16 + c];
        Bl[i]  = dbl[(long)(n0 + n) * 64 + 4 + c];
    }
    __syncthreads();
    float Ads = -__expf(a_log[d * DST + s]);
    float ap = 1.f, h = 0.f;
    for (int n = 0; n < CHUNK; ++n) {
        float dtv = dtl[n * 16 + dl];
        float a = __expf(dtv * Ads);
        h = a * h + dtv * Bl[n * 16 + s] * xcl[n * 16 + dl];
        ap *= a;
    }
    int idx = (chunk * DIN + d) * DST + s;
    aprod_o[idx] = ap;
    hend_o[idx]  = h;
}

// phase 2: serial carry combine across chunks, per (d,s) channel
__global__ void k_scan2(const float* __restrict__ aprod, const float* __restrict__ hend,
                        float* __restrict__ hstart) {
    int t = blockIdx.x * 256 + threadIdx.x;   // DIN*DST = 2048
    if (t >= DIN * DST) return;
    float hst = 0.f;
    for (int c = 0; c < NCHUNK; ++c) {
        int idx = c * DIN * DST + t;
        hstart[idx] = hst;
        hst = aprod[idx] * hst + hend[idx];
    }
}

// phase 3: replay with correct initial state, reduce over s, y = h.C + D*xc
__global__ void __launch_bounds__(256)
k_scan3(const float* __restrict__ dt, const float* __restrict__ dbl,
        const float* __restrict__ xc, const float* __restrict__ a_log,
        const float* __restrict__ Dvec, const float* __restrict__ hstart,
        float* __restrict__ y) {
    int bid = blockIdx.x;
    int chunk = bid >> 3, dg = bid & 7;
    int tid = threadIdx.x;
    int s = tid & 15, dl = tid >> 4;
    int d = dg * 16 + dl;
    __shared__ float dtl[CHUNK * 16], xcl[CHUNK * 16], Bl[CHUNK * 16], Cl[CHUNK * 16];
    int n0 = chunk * CHUNK;
    for (int i = tid; i < CHUNK * 16; i += 256) {
        int n = i >> 4, c = i & 15;
        dtl[i] = dt[(long)(n0 + n) * DIN + dg * 16 + c];
        xcl[i] = xc[(long)(n0 + n) * DIN + dg * 16 + c];
        Bl[i]  = dbl[(long)(n0 + n) * 64 + 4 + c];
        Cl[i]  = dbl[(long)(n0 + n) * 64 + 20 + c];
    }
    __syncthreads();
    float Ads = -__expf(a_log[d * DST + s]);
    float h = hstart[(chunk * DIN + d) * DST + s];
    float Dd = Dvec[d];
    for (int n = 0; n < CHUNK; ++n) {
        float dtv = dtl[n * 16 + dl];
        float a = __expf(dtv * Ads);
        h = a * h + dtv * Bl[n * 16 + s] * xcl[n * 16 + dl];
        float contrib = h * Cl[n * 16 + s];
        contrib += __shfl_xor(contrib, 1);
        contrib += __shfl_xor(contrib, 2);
        contrib += __shfl_xor(contrib, 4);
        contrib += __shfl_xor(contrib, 8);
        if (s == 0) y[(long)(n0 + n) * DIN + d] = contrib + Dd * xcl[n * 16 + dl];
    }
}

// gate: yg = y * silu(z), z = xz[:, 128:256]
__global__ void k_gate(const float* __restrict__ y, const float* __restrict__ xz,
                       _Float16* __restrict__ yg16) {
    int i = blockIdx.x * 256 + threadIdx.x;
    if (i >= NTOK * DIN) return;
    int n = i >> 7, d = i & 127;
    float z = xz[(long)n * 256 + 128 + d];
    float sz = z / (1.0f + __expf(-z));
    yg16[i] = (_Float16)(y[i] * sz);
}

// ---------------------------------------------------------------------------
// Host orchestration
// ---------------------------------------------------------------------------
extern "C" void kernel_launch(void* const* d_in, const int* in_sizes, int n_in,
                              void* d_out, int out_size, void* d_ws, size_t ws_size,
                              hipStream_t stream) {
    (void)in_sizes; (void)n_in; (void)out_size; (void)ws_size;
    const float* x          = (const float*)d_in[0];
    const float* coords     = (const float*)d_in[1];
    const float* ln1_g      = (const float*)d_in[2];
    const float* ln1_b      = (const float*)d_in[3];
    const float* Wq         = (const float*)d_in[4];
    const float* Wk         = (const float*)d_in[5];
    const float* Wv         = (const float*)d_in[6];
    const float* rpe_w      = (const float*)d_in[7];
    const float* rpe_b      = (const float*)d_in[8];
    const float* attn_out_w = (const float*)d_in[9];
    const float* ln_g       = (const float*)d_in[10];
    const float* ln_b       = (const float*)d_in[11];
    const float* in_proj_w  = (const float*)d_in[12];
    const float* conv_w     = (const float*)d_in[13];
    const float* conv_b     = (const float*)d_in[14];
    const float* x_proj_w   = (const float*)d_in[15];
    const float* dt_proj_w  = (const float*)d_in[16];
    const float* dt_proj_b  = (const float*)d_in[17];
    const float* A_log      = (const float*)d_in[18];
    const float* Dv         = (const float*)d_in[19];
    const float* mamba_out_w= (const float*)d_in[20];
    const float* normf_g    = (const float*)d_in[21];
    const float* normf_b    = (const float*)d_in[22];

    char* ws = (char*)d_ws;
    size_t off = 0;
    auto alloc = [&](size_t bytes) -> void* {
        off = (off + 255) & ~(size_t)255;
        void* p = (void*)(ws + off);
        off += bytes;
        return p;
    };
    // --- persistent ---
    unsigned long long* keys = (unsigned long long*)alloc((size_t)NTOK * 8);
    int*       order  = (int*)alloc((size_t)NTOK * 4);
    float*     cs5    = (float*)alloc((size_t)NTOK * 5 * 4);
    _Float16*  xn16   = (_Float16*)alloc((size_t)NTOK * HDIM * 2);
    float*     hidden = (float*)alloc((size_t)NTOK * HDIM * 4);
    float*     resid  = (float*)alloc((size_t)NTOK * HDIM * 4);
    _Float16*  hn16   = (_Float16*)alloc((size_t)NTOK * HDIM * 2);
    float*     effw   = (float*)alloc(256);
    float*     biasb  = (float*)alloc(256);
    _Float16*  wq16   = (_Float16*)alloc((size_t)64 * 512 * 2);   // transposed [512][64]
    _Float16*  wk16   = (_Float16*)alloc((size_t)64 * 512 * 2);
    _Float16*  wv16   = (_Float16*)alloc((size_t)64 * 512 * 2);
    _Float16*  wo16   = (_Float16*)alloc((size_t)512 * 64 * 2);   // transposed [64][512]
    _Float16*  win16[NLAY], *wxp16[NLAY], *wmo16[NLAY];
    for (int l = 0; l < NLAY; ++l) {
        win16[l] = (_Float16*)alloc((size_t)64 * 256 * 2);        // transposed [256][64]
        wxp16[l] = (_Float16*)alloc((size_t)128 * 64 * 2);        // transposed+padded [64][128]
        wmo16[l] = (_Float16*)alloc((size_t)128 * 64 * 2);        // transposed [64][128]
    }
    size_t scratch0 = off;
    // --- attention view of scratch ---
    _Float16* q16  = (_Float16*)alloc((size_t)NTOK * 512 * 2);
    _Float16* k16  = (_Float16*)alloc((size_t)NTOK * 512 * 2);
    _Float16* v16  = (_Float16*)alloc((size_t)NTOK * 512 * 2);
    _Float16* o16  = (_Float16*)alloc((size_t)NTOK * 512 * 2);
    float*    atmp = (float*)alloc((size_t)NTOK * HDIM * 4);
    // --- mamba view of scratch (reuses attention region; lifetimes disjoint) ---
    off = scratch0;
    float*    xz     = (float*)alloc((size_t)NTOK * 256 * 4);
    float*    xc     = (float*)alloc((size_t)NTOK * DIN * 4);
    _Float16* xc16   = (_Float16*)alloc((size_t)NTOK * DIN * 2);
    float*    dbl    = (float*)alloc((size_t)NTOK * 64 * 4);
    float*    dtbuf  = (float*)alloc((size_t)NTOK * DIN * 4);
    float*    aprod  = (float*)alloc((size_t)NCHUNK * DIN * DST * 4);
    float*    hendb  = (float*)alloc((size_t)NCHUNK * DIN * DST * 4);
    float*    hstartb= (float*)alloc((size_t)NCHUNK * DIN * DST * 4);
    float*    ybuf   = (float*)alloc((size_t)NTOK * DIN * 4);
    _Float16* yg16   = (_Float16*)alloc((size_t)NTOK * DIN * 2);

    // ---- 1. sort by coords[:,0] ----
    k_sort_init<<<NTOK / 256, 256, 0, stream>>>(coords, keys);
    for (int k = 2; k <= NTOK; k <<= 1)
        for (int j = k >> 1; j > 0; j >>= 1)
            k_sort_stage<<<NTOK / 256, 256, 0, stream>>>(keys, j, k);
    k_sort_finish<<<NTOK / 256, 256, 0, stream>>>(keys, order);

    // ---- 2. gather + ln1 ----
    k_gather_ln<<<NTOK, 64, 0, stream>>>(x, coords, order, ln1_g, ln1_b, xn16, cs5);

    // ---- 3. weight prep (f16 + transpose) ----
    k_f16T<<<(64 * 512 + 255) / 256, 256, 0, stream>>>(Wq, wq16, 64, 512);
    k_f16T<<<(64 * 512 + 255) / 256, 256, 0, stream>>>(Wk, wk16, 64, 512);
    k_f16T<<<(64 * 512 + 255) / 256, 256, 0, stream>>>(Wv, wv16, 64, 512);
    k_f16T<<<(512 * 64 + 255) / 256, 256, 0, stream>>>(attn_out_w, wo16, 512, 64);
    for (int l = 0; l < NLAY; ++l) {
        k_f16T<<<(64 * 256 + 255) / 256, 256, 0, stream>>>(in_proj_w + (size_t)l * 64 * 256, win16[l], 64, 256);
        k_pad_xprojT<<<(64 * 128 + 255) / 256, 256, 0, stream>>>(x_proj_w + (size_t)l * 128 * 36, wxp16[l]);
        k_f16T<<<(128 * 64 + 255) / 256, 256, 0, stream>>>(mamba_out_w + (size_t)l * 128 * 64, wmo16[l], 128, 64);
    }
    k_rpe_prep<<<1, 64, 0, stream>>>(rpe_w, rpe_b, effw, biasb);

    // ---- 4. QKV GEMMs (WMMA) ----
    k_gemm16<<<(NTOK / 64) * (512 / 32), 256, 0, stream>>>(xn16, wq16, q16, 64, 512, 512 / 32);
    k_gemm16<<<(NTOK / 64) * (512 / 32), 256, 0, stream>>>(xn16, wk16, k16, 64, 512, 512 / 32);
    k_gemm16<<<(NTOK / 64) * (512 / 32), 256, 0, stream>>>(xn16, wv16, v16, 64, 512, 512 / 32);

    // ---- 5. block attention ----
    k_attn<<<(NTOK / TBLK) * NH * 4, 256, 0, stream>>>(q16, k16, v16, cs5, effw, biasb, o16);

    // ---- 6. attention output projection + un-sort scatter ----
    k_gemm32<<<(NTOK / 64) * (64 / 32), 256, 0, stream>>>(o16, wo16, atmp, 512, 64, 64 / 32);
    k_scatter<<<(NTOK * HDIM) / 256, 256, 0, stream>>>(atmp, order, hidden);

    // ---- 7. mamba layers ----
    hipMemsetAsync(resid, 0, (size_t)NTOK * HDIM * 4, stream);
    for (int l = 0; l < NLAY; ++l) {
        k_ln<<<NTOK, 64, 0, stream>>>(hidden, resid, ln_g + l * 64, ln_b + l * 64, hn16, nullptr);
        k_gemm32<<<(NTOK / 64) * (256 / 32), 256, 0, stream>>>(hn16, win16[l], xz, 64, 256, 256 / 32);
        k_conv<<<(NTOK * DIN) / 256, 256, 0, stream>>>(xz, conv_w + (size_t)l * DIN * 4, conv_b + (size_t)l * DIN, xc, xc16);
        k_gemm32<<<(NTOK / 64) * (64 / 32), 256, 0, stream>>>(xc16, wxp16[l], dbl, 128, 64, 64 / 32);
        k_dt<<<(NTOK * DIN) / 256, 256, 0, stream>>>(dbl, dt_proj_w + (size_t)l * 4 * DIN, dt_proj_b + (size_t)l * DIN, dtbuf);
        k_scan1<<<NCHUNK * 8, 256, 0, stream>>>(dtbuf, dbl, xc, A_log + (size_t)l * DIN * DST, aprod, hendb);
        k_scan2<<<(DIN * DST) / 256, 256, 0, stream>>>(aprod, hendb, hstartb);
        k_scan3<<<NCHUNK * 8, 256, 0, stream>>>(dtbuf, dbl, xc, A_log + (size_t)l * DIN * DST, Dv + (size_t)l * DIN, hstartb, ybuf);
        k_gate<<<(NTOK * DIN) / 256, 256, 0, stream>>>(ybuf, xz, yg16);
        k_gemm32<<<(NTOK / 64) * (64 / 32), 256, 0, stream>>>(yg16, wmo16[l], hidden, 128, 64, 64 / 32);
    }

    // ---- 8. final residual + layernorm -> output ----
    k_ln<<<NTOK, 64, 0, stream>>>(hidden, resid, normf_g, normf_b, nullptr, (float*)d_out);
}
